// Instance_83253646066167
// MI455X (gfx1250) — matrix operand order repaired and checked by reading, not compiled, not run on hardware
//
#include <hip/hip_runtime.h>
#include <hip/hip_bf16.h>

typedef __attribute__((ext_vector_type(16))) __bf16 bf16x16;
typedef __attribute__((ext_vector_type(8)))  float  f32x8;

namespace {
constexpr int Bc = 2, Lc = 2048, Dc = 1024, Hc = 16, Dhc = 64, HIDc = 4096;
constexpr int BLc = Bc * Lc;            // 4096 rows (tokens)
constexpr int WINc = 256, DILc = 2;

// GEMM tiling: 256 threads = 8 waves; block tile 128x128; wave tile 32x64; K-stage 64.
constexpr int BM = 128, BN = 128, BK = 64;
constexpr int ASTR = 72;                // LDS row stride (bf16), padded (16B-aligned rows)
constexpr int BSTR = 72;
}

// Low 32 bits of a flat LDS pointer == LDS byte offset (AS3) on amdgcn.
__device__ __forceinline__ uint32_t lds_addr(const void* p) {
  return (uint32_t)(uintptr_t)p;
}

// CDNA5 async global->LDS copies (ASYNCcnt-tracked, §15.18.3).  GV mode:
// copies global vaddr+offset into LDS vdst+offset.  64B per thread.
__device__ __forceinline__ void async_g2l_64B(uint32_t lds, uint64_t g) {
  asm volatile(
      "global_load_async_to_lds_b128 %0, %1, off\n\t"
      "global_load_async_to_lds_b128 %0, %1, off offset:16\n\t"
      "global_load_async_to_lds_b128 %0, %1, off offset:32\n\t"
      "global_load_async_to_lds_b128 %0, %1, off offset:48"
      :: "v"(lds), "v"(g) : "memory");
}
__device__ __forceinline__ void wait_async0() {
  asm volatile("s_wait_asynccnt 0x0" ::: "memory");
}

// ---------------------------------------------------------------------------
// fp32 [K,N] -> bf16 [N,K] via 32x32 LDS tile (coalesced read AND write)
// ---------------------------------------------------------------------------
__global__ __launch_bounds__(256)
void convert_transpose_bf16(const float* __restrict__ W, __bf16* __restrict__ Wt,
                            int K, int N) {
  __shared__ __bf16 t[32][33];
  const int k0 = blockIdx.x * 32, n0 = blockIdx.y * 32;
  const int tx = threadIdx.x & 31, ty = threadIdx.x >> 5;   // 32 x 8
#pragma unroll
  for (int e = 0; e < 4; ++e) {
    const int r = ty + e * 8;
    t[r][tx] = (__bf16)W[(size_t)(k0 + r) * N + n0 + tx];
  }
  __syncthreads();
#pragma unroll
  for (int e = 0; e < 4; ++e) {
    const int r = ty + e * 8;                                // r -> n index
    Wt[(size_t)(n0 + r) * K + k0 + tx] = t[tx][r];
  }
}

// ---------------------------------------------------------------------------
// LayerNorm over D=1024, one block per row, output bf16
// ---------------------------------------------------------------------------
__global__ __launch_bounds__(256)
void layernorm_to_bf16(const float* __restrict__ x, const float* __restrict__ w,
                       const float* __restrict__ b, __bf16* __restrict__ out) {
  const int row = blockIdx.x;
  const float* xr = x + (size_t)row * Dc;
  float v4[4];
  float s = 0.f, s2 = 0.f;
#pragma unroll
  for (int e = 0; e < 4; ++e) {
    float t = xr[threadIdx.x + e * 256];
    v4[e] = t; s += t; s2 += t * t;
  }
#pragma unroll
  for (int off = 16; off > 0; off >>= 1) {
    s  += __shfl_xor(s, off, 32);
    s2 += __shfl_xor(s2, off, 32);
  }
  __shared__ float red[16];
  const int lane = threadIdx.x & 31, wv = threadIdx.x >> 5;
  if (lane == 0) { red[wv] = s; red[wv + 8] = s2; }
  __syncthreads();
  float S = 0.f, S2 = 0.f;
#pragma unroll
  for (int i = 0; i < 8; ++i) { S += red[i]; S2 += red[i + 8]; }
  const float mean = S * (1.0f / Dc);
  const float var  = S2 * (1.0f / Dc) - mean * mean;
  const float rstd = rsqrtf(var + 1e-5f);
  __bf16* orow = out + (size_t)row * Dc;
#pragma unroll
  for (int e = 0; e < 4; ++e) {
    int idx = threadIdx.x + e * 256;
    orow[idx] = (__bf16)((v4[e] - mean) * rstd * w[idx] + b[idx]);
  }
}

// ---------------------------------------------------------------------------
// Dilated-window attention, fp32, one wave per (b,h,query); lanes <-> dims.
// ---------------------------------------------------------------------------
__global__ __launch_bounds__(256)
void attn_dilated(const float* __restrict__ q, const float* __restrict__ k,
                  const float* __restrict__ v, __bf16* __restrict__ obf) {
  const int lane = threadIdx.x & 31;
  const int wid  = blockIdx.x * 8 + (threadIdx.x >> 5);   // 0 .. B*H*L-1
  const int i  = wid & (Lc - 1);
  const int bh = wid >> 11;                               // b*H + h
  const float* qr = q + ((size_t)bh * Lc + i) * Dhc;
  const float* kb = k + (size_t)bh * Lc * Dhc;
  const float* vb = v + (size_t)bh * Lc * Dhc;
  const float q0 = qr[lane], q1 = qr[lane + 32];
  const float scale = 0.125f;                             // 1/sqrt(64)
  float mrun = -3.0e38f, lrun = 0.f, o0 = 0.f, o1 = 0.f;
  const int tmax = min(WINc / DILc, i >> 1);
  for (int t = 0; t <= tmax; ++t) {
    const int j = i - DILc * t;
    const float* kr = kb + (size_t)j * Dhc;
    float s = q0 * kr[lane] + q1 * kr[lane + 32];
#pragma unroll
    for (int off = 16; off > 0; off >>= 1) s += __shfl_xor(s, off, 32);
    s *= scale;
    const float mnew = fmaxf(mrun, s);
    const float corr = __expf(mrun - mnew);
    const float p    = __expf(s - mnew);
    const float* vr  = vb + (size_t)j * Dhc;
    lrun = lrun * corr + p;
    o0   = o0 * corr + p * vr[lane];
    o1   = o1 * corr + p * vr[lane + 32];
    mrun = mnew;
  }
  const float inv = 1.0f / lrun;
  const int b = bh >> 4, h = bh & 15;
  __bf16* orow = obf + ((size_t)(b * Lc + i) * Hc + h) * Dhc;
  orow[lane]      = (__bf16)(o0 * inv);
  orow[lane + 32] = (__bf16)(o1 * inv);
}

// ---------------------------------------------------------------------------
// bf16 WMMA GEMM with async-LDS double buffering:
//   C[M,N] = A[M,K] * Bt[N,K]^T  (+ fused epilogue)
// MODE 0: QKV  -> scatter fp32 into q/k/v [B,H,L,Dh] (+bias)
// MODE 1: OUT  -> fp32 out = acc + bias + resid
// MODE 2: FFN1 -> bf16 out = gelu(acc + bias), row stride HID
// MODE 3: FFN2 -> fp32 out = acc + bias + resid
// ---------------------------------------------------------------------------
template <int MODE>
__global__ __launch_bounds__(256)
void gemm_bf16_wmma(const __bf16* __restrict__ A, const __bf16* __restrict__ Bt,
                    const float* __restrict__ bias, const float* __restrict__ resid,
                    float* __restrict__ outF, __bf16* __restrict__ outBF, int K) {
  __shared__ __bf16 As[2][BM * ASTR];
  __shared__ __bf16 Bs[2][BN * BSTR];

  const int tid  = threadIdx.x;
  const int lane = tid & 31;
  const int wave = tid >> 5;
  const int wm = wave >> 1;            // 0..3 -> M sub-tile (32 rows)
  const int wn = wave & 1;             // 0..1 -> N sub-tile (64 cols)
  const int m_blk = blockIdx.x * BM;
  const int n_blk = blockIdx.y * BN;

  f32x8 acc[2][4] = {};

  // Staging: A tile 128x64 and B tile 128x64 bf16 -> 64B/thread each.
  const int srow = tid >> 1, scol = (tid & 1) * 32;
  const __bf16* aSrc = A  + (size_t)(m_blk + srow) * K + scol;
  const __bf16* bSrc = Bt + (size_t)(n_blk + srow) * K + scol;
  const uint32_t aL[2] = { lds_addr(&As[0][srow * ASTR + scol]),
                           lds_addr(&As[1][srow * ASTR + scol]) };
  const uint32_t bL[2] = { lds_addr(&Bs[0][srow * BSTR + scol]),
                           lds_addr(&Bs[1][srow * BSTR + scol]) };

  const int nst = K / BK;
  // prologue: stage 0 into buffer 0
  async_g2l_64B(aL[0], (uint64_t)(uintptr_t)aSrc);
  async_g2l_64B(bL[0], (uint64_t)(uintptr_t)bSrc);

  for (int it = 0; it < nst; ++it) {
    wait_async0();
    __syncthreads();                       // buffer it&1 now valid for all waves
    if (it + 1 < nst) {                    // prefetch next stage into other buffer
      const int kt = (it + 1) * BK;
      async_g2l_64B(aL[(it + 1) & 1], (uint64_t)(uintptr_t)(aSrc + kt));
      async_g2l_64B(bL[(it + 1) & 1], (uint64_t)(uintptr_t)(bSrc + kt));
    }
    const __bf16* Ab = As[it & 1];
    const __bf16* Bb = Bs[it & 1];

#pragma unroll
    for (int ks = 0; ks < BK; ks += 32) {
      // A fragment (ISA 16-bit A 16x32 layout):
      //   lane<16: V0..3=K0..7, V4..7=K16..23 ; lanes>=16: +8
      bf16x16 af[2];
      const int g8 = (lane & 16) ? 8 : 0;
#pragma unroll
      for (int mf = 0; mf < 2; ++mf) {
        const int row = wm * 32 + mf * 16 + (lane & 15);
        union { unsigned u[8]; bf16x16 vv; } tu;
#pragma unroll
        for (int pr = 0; pr < 8; ++pr) {
          const int kk = ks + ((pr & 3) << 1) + ((pr >= 4) ? 16 : 0) + g8;
          tu.u[pr] = *(const unsigned*)&Ab[row * ASTR + kk];
        }
        af[mf] = tu.vv;
      }
      // B fragment: lane&15 = N; lanes 0-15 K0..7,K16..23; lanes 16-31 K8..15,K24..31
      bf16x16 bfr[4];
      const int kbo = ks + ((lane & 16) ? 8 : 0);
#pragma unroll
      for (int nf = 0; nf < 4; ++nf) {
        const int row = wn * 64 + nf * 16 + (lane & 15);
        union { uint4 qv[2]; bf16x16 vv; } tu;
        const uint4* pq = (const uint4*)&Bb[row * BSTR + kbo];
        tu.qv[0] = pq[0]; tu.qv[1] = pq[2];
        bfr[nf] = tu.vv;
      }
#pragma unroll
      for (int mf = 0; mf < 2; ++mf)
#pragma unroll
        for (int nf = 0; nf < 4; ++nf)
          acc[mf][nf] = __builtin_amdgcn_wmma_f32_16x16x32_bf16(
              false, af[mf], false, bfr[nf], (short)0, acc[mf][nf], false, false);
    }
  }

  // Epilogue. C/D layout: VGPR r -> M = r (+8 for lanes>=16), lane&15 -> N.
#pragma unroll
  for (int mf = 0; mf < 2; ++mf)
#pragma unroll
    for (int nf = 0; nf < 4; ++nf) {
      const int n  = n_blk + wn * 64 + nf * 16 + (lane & 15);
      const int mb = m_blk + wm * 32 + mf * 16 + ((lane & 16) ? 8 : 0);
      const float bn = bias[n];
#pragma unroll
      for (int r = 0; r < 8; ++r) {
        const int m = mb + r;
        float val = acc[mf][nf][r] + bn;
        if (MODE == 0) {
          const int part = n >> 10;              // 0=q 1=k 2=v
          const int d    = n & 1023;
          const int head = d >> 6, dh = d & 63;
          const int bh   = ((m >> 11) << 4) + head;   // b*H + head
          const int l    = m & 2047;
          outF[(size_t)part * ((size_t)BLc * Dc) +
               ((size_t)bh * Lc + l) * Dhc + dh] = val;
        } else if (MODE == 2) {
          const float g = 0.5f * val * (1.0f + erff(val * 0.70710678118654752f));
          outBF[(size_t)m * HIDc + n] = (__bf16)g;
        } else {  // MODE 1 / 3: bias + residual, fp32
          outF[(size_t)m * Dc + n] = val + resid[(size_t)m * Dc + n];
        }
      }
    }
}

// ---------------------------------------------------------------------------
extern "C" void kernel_launch(void* const* d_in, const int* in_sizes, int n_in,
                              void* d_out, int out_size, void* d_ws, size_t ws_size,
                              hipStream_t stream) {
  (void)in_sizes; (void)n_in; (void)out_size; (void)ws_size;
  const float* x      = (const float*)d_in[0];
  const float* n1w    = (const float*)d_in[1];
  const float* n1b    = (const float*)d_in[2];
  const float* qkv_w  = (const float*)d_in[3];
  const float* qkv_b  = (const float*)d_in[4];
  const float* out_w  = (const float*)d_in[5];
  const float* out_b  = (const float*)d_in[6];
  const float* n2w    = (const float*)d_in[7];
  const float* n2b    = (const float*)d_in[8];
  const float* ffn_w1 = (const float*)d_in[9];
  const float* ffn_b1 = (const float*)d_in[10];
  const float* ffn_w2 = (const float*)d_in[11];
  const float* ffn_b2 = (const float*)d_in[12];

  char* p = (char*)d_ws;
  __bf16* xn   = (__bf16*)p;  p += (size_t)BLc * Dc * 2;          //  8 MB LN1(x) bf16
  float*  qkvf = (float*)p;   p += (size_t)3 * BLc * Dc * 4;      // 48 MB q,k,v [B,H,L,Dh]
  __bf16* obf  = (__bf16*)p;  p += (size_t)BLc * Dc * 2;          //  8 MB attn out bf16
  float*  y    = (float*)p;   p += (size_t)BLc * Dc * 4;          // 16 MB residual-1
  __bf16* hbf  = (__bf16*)p;  p += (size_t)BLc * Dc * 2;          //  8 MB LN2(y) bf16
  __bf16* h1   = (__bf16*)p;  p += (size_t)BLc * HIDc * 2;        // 32 MB gelu out bf16
  __bf16* wqkv = (__bf16*)p;  p += (size_t)Dc * 3 * Dc * 2;       //  6 MB
  __bf16* wout = (__bf16*)p;  p += (size_t)Dc * Dc * 2;           //  2 MB
  __bf16* w1t  = (__bf16*)p;  p += (size_t)Dc * HIDc * 2;         //  8 MB
  __bf16* w2t  = (__bf16*)p;  p += (size_t)HIDc * Dc * 2;         //  8 MB

  // Weights fp32[K,N] -> bf16[N,K] (tiled transpose)
  convert_transpose_bf16<<<dim3(Dc / 32, 3 * Dc / 32), 256, 0, stream>>>(qkv_w, wqkv, Dc, 3 * Dc);
  convert_transpose_bf16<<<dim3(Dc / 32, Dc / 32),     256, 0, stream>>>(out_w, wout, Dc, Dc);
  convert_transpose_bf16<<<dim3(Dc / 32, HIDc / 32),   256, 0, stream>>>(ffn_w1, w1t, Dc, HIDc);
  convert_transpose_bf16<<<dim3(HIDc / 32, Dc / 32),   256, 0, stream>>>(ffn_w2, w2t, HIDc, Dc);

  layernorm_to_bf16<<<BLc, 256, 0, stream>>>(x, n1w, n1b, xn);

  gemm_bf16_wmma<0><<<dim3(BLc / BM, 3 * Dc / BN), 256, 0, stream>>>(
      xn, wqkv, qkv_b, nullptr, qkvf, nullptr, Dc);

  const float* qf = qkvf;
  const float* kf = qkvf + (size_t)BLc * Dc;
  const float* vf = qkvf + (size_t)2 * BLc * Dc;
  attn_dilated<<<(Bc * Hc * Lc) / 8, 256, 0, stream>>>(qf, kf, vf, obf);

  gemm_bf16_wmma<1><<<dim3(BLc / BM, Dc / BN), 256, 0, stream>>>(
      obf, wout, out_b, x, y, nullptr, Dc);

  layernorm_to_bf16<<<BLc, 256, 0, stream>>>(y, n2w, n2b, hbf);

  gemm_bf16_wmma<2><<<dim3(BLc / BM, HIDc / BN), 256, 0, stream>>>(
      hbf, w1t, ffn_b1, nullptr, nullptr, h1, Dc);

  gemm_bf16_wmma<3><<<dim3(BLc / BM, Dc / BN), 256, 0, stream>>>(
      h1, w2t, ffn_b2, y, (float*)d_out, nullptr, HIDc);
}
